// TSModel_47949014892893
// MI455X (gfx1250) — compile-verified, matrix operand order checked by reading, not run on hardware
//
#include <hip/hip_runtime.h>
#include <stdint.h>

// LSTM: B=256, T=512, D=1, H=512.  Gate order (i, j, f, o), forget bias = 1.0.
// Per-timestep bf16 WMMA GEMM h[256x512] @ Wh[512x2048], fp32 accumulators,
// gates fused in the epilogue. Wh is pre-packed once into the CDNA5 B-matrix
// fragment layout. Each wave computes a 32(M) x 16(N-of-H) tile for all four
// gates (8 WMMA tiles). The shared 32x512 h tile is staged once per block in
// LDS (8 waves share it); weight fragments stream from L2 software-pipelined
// so loads for k+1 overlap the WMMAs of k.

typedef __attribute__((ext_vector_type(16))) __bf16 v16bf;
typedef __attribute__((ext_vector_type(8)))  float  v8f;

#define B_   256
#define T_   512
#define H_   512
#define K_   512      // recurrent GEMM depth (= H)
#define G4_  2048     // 4*H

__device__ __forceinline__ unsigned short f2bf(float f) {
    unsigned int u = __float_as_uint(f);
    u += 0x7FFFu + ((u >> 16) & 1u);           // round-to-nearest-even
    return (unsigned short)(u >> 16);
}
__device__ __forceinline__ float sigmoid_fast(float v) {
    return 1.0f / (1.0f + __expf(-v));
}
__device__ __forceinline__ float tanh_fast(float v) {
    float e = __expf(-2.0f * v);
    return (1.0f - e) / (1.0f + e);
}

// 32 contiguous bytes -> one bf16 B-fragment register block
__device__ __forceinline__ v16bf load_fragB(const unsigned short* p) {
    v16bf v;
    ((uint4*)&v)[0] = ((const uint4*)p)[0];
    ((uint4*)&v)[1] = ((const uint4*)p)[1];
    return v;
}
// A fragment from row-major bf16: 16B at +0 elements, 16B at +16 elements
__device__ __forceinline__ v16bf load_fragA(const unsigned short* p) {
    v16bf v;
    ((uint4*)&v)[0] = ((const uint4*)p)[0];
    ((uint4*)&v)[1] = ((const uint4*)p)[2];
    return v;
}

// ---------------------------------------------------------------------------
// Pre-pack Wh (rows 1..512 of W, fp32) into bf16 WMMA B-fragment layout.
// Fragment (nt, kt): 32x16 bf16 tile of z-cols [nt*16, nt*16+16) and K rows
// [kt*32, kt*32+32). 16 bf16 per lane, lane-major:
//   lane l<16 : col n = nt*16+l ;  e=0..7 -> K=k0+e,    e=8..15 -> K=k0+16+(e-8)
//   lane l>=16: col n = nt*16+l-16; e=0..7 -> K=k0+8+e, e=8..15 -> K=k0+24+(e-8)
// ---------------------------------------------------------------------------
__global__ void lstm_prepack_w(const float* __restrict__ W,
                               unsigned short* __restrict__ Wp) {
    unsigned int gid  = blockIdx.x * blockDim.x + threadIdx.x;   // 0 .. 2^20-1
    unsigned int e    = gid & 15u;
    unsigned int lane = (gid >> 4) & 31u;
    unsigned int kt   = (gid >> 9) & 15u;
    unsigned int nt   = gid >> 13;                                // 0..127
    unsigned int n    = nt * 16u + (lane & 15u);
    unsigned int hi   = lane >> 4;
    unsigned int kk   = (e < 8u) ? (e + hi * 8u) : (16u + (e - 8u) + hi * 8u);
    unsigned int k    = kt * 32u + kk;
    Wp[gid] = f2bf(W[(size_t)(1u + k) * G4_ + n]);  // skip row 0 (x weights)
}

// ---------------------------------------------------------------------------
// One LSTM timestep. Grid (8, 4), block 256 (8 waves).
// Wave w of block (bx, by): rows m in [bx*32, bx*32+32),
// H-cols n in [by*128 + w*16, ..+16), all four gates.
// ---------------------------------------------------------------------------
__global__ __launch_bounds__(256) void lstm_step(
    const float* __restrict__ x,            // [B, T] (D=1)
    const float* __restrict__ W,            // [513, 2048] fp32 (row 0 = Wx)
    const float* __restrict__ bias,         // [2048]
    const unsigned short* __restrict__ Wp,  // packed bf16 Wh fragments
    const unsigned short* __restrict__ h_in,   // [B, H] bf16
    unsigned short* __restrict__ h_out,        // [B, H] bf16
    float* __restrict__ c_state,               // [B, H] fp32
    float* __restrict__ out,                   // [B, T, H] fp32
    int t)
{
    __shared__ unsigned short sh[32 * K_];     // 32 KB: h rows [m0, m0+32)

    const unsigned int lane  = threadIdx.x & 31u;
    const unsigned int wave  = threadIdx.x >> 5;
    const unsigned int m0    = blockIdx.x * 32u;
    const unsigned int n0    = blockIdx.y * 128u + wave * 16u;
    const unsigned int nt_h  = n0 >> 4;            // H col-tile index 0..31
    const unsigned int laneN = lane & 15u;
    const unsigned int hi    = lane >> 4;

    // ---- stage shared A tile: rows are contiguous -> flat 32 KB copy ----
    {
        const uint4* src = (const uint4*)(h_in + (size_t)m0 * K_);
        uint4* dst = (uint4*)sh;
#pragma unroll
        for (int i = 0; i < 8; ++i)
            dst[threadIdx.x + i * 256] = src[threadIdx.x + i * 256];
    }
    __syncthreads();

    v8f acc[4][2];
#pragma unroll
    for (int g = 0; g < 4; ++g)
#pragma unroll
        for (int mt = 0; mt < 2; ++mt)
            acc[g][mt] = (v8f){0.f,0.f,0.f,0.f,0.f,0.f,0.f,0.f};

    // A-fragment lane base pointers into LDS (local rows 0..31)
    const unsigned short* aptr0 = sh + (size_t)laneN * K_ + hi * 8u;
    const unsigned short* aptr1 = aptr0 + 16u * K_;
    // B-fragment lane base pointer for gate g at k-tile kt:
    //   Wp + ((g*32 + nt_h)*16 + kt)*512 + lane*16
    const unsigned short* bbase = Wp + (size_t)nt_h * 16u * 512u + lane * 16u;

    // ---- software-pipelined K loop: fetch (kt+1) while multiplying kt ----
    v16bf Acur0 = load_fragA(aptr0);
    v16bf Acur1 = load_fragA(aptr1);
    v16bf Bcur[4];
#pragma unroll
    for (int g = 0; g < 4; ++g)
        Bcur[g] = load_fragB(bbase + (size_t)g * 32u * 16u * 512u);

#pragma unroll
    for (unsigned int kt = 0; kt < 16u; ++kt) {
        v16bf Anxt0, Anxt1, Bnxt[4];
        if (kt < 15u) {
            Anxt0 = load_fragA(aptr0 + (kt + 1u) * 32u);
            Anxt1 = load_fragA(aptr1 + (kt + 1u) * 32u);
#pragma unroll
            for (int g = 0; g < 4; ++g)
                Bnxt[g] = load_fragB(bbase + (size_t)g * 32u * 16u * 512u
                                           + (kt + 1u) * 512u);
        }
#pragma unroll
        for (int g = 0; g < 4; ++g) {
            acc[g][0] = __builtin_amdgcn_wmma_f32_16x16x32_bf16(
                false, Acur0, false, Bcur[g], (short)0, acc[g][0], false, false);
            acc[g][1] = __builtin_amdgcn_wmma_f32_16x16x32_bf16(
                false, Acur1, false, Bcur[g], (short)0, acc[g][1], false, false);
        }
        if (kt < 15u) {
            Acur0 = Anxt0;
            Acur1 = Anxt1;
#pragma unroll
            for (int g = 0; g < 4; ++g) Bcur[g] = Bnxt[g];
        }
    }

    // ---------------- epilogue: rank-1 x term + bias + gates -------------
    const unsigned int n = n0 + laneN;
    float wxv[4], bv[4];
#pragma unroll
    for (int g = 0; g < 4; ++g) {
        wxv[g] = W[g * H_ + n];        // W row 0 (x weight), z-col g*512+n
        bv[g]  = bias[g * H_ + n];
    }

#pragma unroll
    for (int mt = 0; mt < 2; ++mt) {
#pragma unroll
        for (int r = 0; r < 8; ++r) {
            const unsigned int m =
                m0 + (unsigned int)mt * 16u + hi * 8u + (unsigned int)r;
            const float xv = x[(size_t)m * T_ + (unsigned int)t];

            float zi = acc[0][mt][r] + xv * wxv[0] + bv[0];
            float zj = acc[1][mt][r] + xv * wxv[1] + bv[1];
            float zf = acc[2][mt][r] + xv * wxv[2] + bv[2];
            float zo = acc[3][mt][r] + xv * wxv[3] + bv[3];

            float ig = sigmoid_fast(zi);
            float jg = tanh_fast(zj);
            float fg = sigmoid_fast(zf + 1.0f);   // forget bias
            float og = sigmoid_fast(zo);

            const unsigned int cidx = m * H_ + n;
            float cn = c_state[cidx] * fg + ig * jg;
            float hn = tanh_fast(cn) * og;

            c_state[cidx] = cn;
            h_out[cidx]   = f2bf(hn);
            out[((size_t)m * T_ + (unsigned int)t) * H_ + n] = hn;
        }
    }
}

// ---------------------------------------------------------------------------
// Workspace layout (bytes):
//   [0,        2 MiB)  : packed Wh bf16 fragments  (128*16*32*16 elems)
//   [2 MiB,  +256 KiB) : h ping  (bf16 [256,512])
//   [.. ,    +256 KiB) : h pong
//   [.. ,    +512 KiB) : c state (fp32 [256,512])
// ---------------------------------------------------------------------------
extern "C" void kernel_launch(void* const* d_in, const int* in_sizes, int n_in,
                              void* d_out, int out_size, void* d_ws, size_t ws_size,
                              hipStream_t stream) {
    (void)in_sizes; (void)n_in; (void)out_size; (void)ws_size;
    const float* x  = (const float*)d_in[0];   // [256, 512, 1]
    const float* W  = (const float*)d_in[1];   // [513, 2048]
    const float* b  = (const float*)d_in[2];   // [2048]
    float* out      = (float*)d_out;           // [256, 512, 512]

    char* ws = (char*)d_ws;
    unsigned short* Wp = (unsigned short*)ws;
    unsigned short* h0 = (unsigned short*)(ws + (2u << 20));
    unsigned short* h1 = h0 + (size_t)B_ * H_;
    float* c           = (float*)(ws + (2u << 20) + (512u << 10));

    // zero initial state every call (deterministic; graph-capturable)
    hipMemsetAsync(h0, 0, (size_t)B_ * H_ * sizeof(unsigned short), stream);
    hipMemsetAsync(c,  0, (size_t)B_ * H_ * sizeof(float), stream);

    // pack Wh into WMMA fragment layout (2^20 elements)
    lstm_prepack_w<<<4096, 256, 0, stream>>>(W, Wp);

    // sequential scan over time
    for (int t = 0; t < T_; ++t) {
        unsigned short* hin  = (t & 1) ? h1 : h0;
        unsigned short* hout = (t & 1) ? h0 : h1;
        lstm_step<<<dim3(8, 4), 256, 0, stream>>>(
            x, W, b, Wp, hin, hout, c, out, t);
    }
}
